// DispersionLoss_5849745457954
// MI455X (gfx1250) — matrix operand order repaired and checked by reading, not compiled, not run on hardware
//
#include <hip/hip_runtime.h>
#include <math.h>

typedef __attribute__((ext_vector_type(16))) __bf16    v16bf;
typedef __attribute__((ext_vector_type(2)))  __bf16    bf16x2;
typedef __attribute__((ext_vector_type(8)))  float     v8f;
typedef __attribute__((ext_vector_type(4)))  float     v4f;
typedef __attribute__((ext_vector_type(8)))  unsigned  v8u;

#define N_PTS   16384
#define F_DIM   64
#define K_BINS  32
#define C_DIM   128
#define FK      (F_DIM * K_BINS)      // 2048
#define NCOLS   144                   // 128 teacher + ysq + ones + 14 pad
#define NTILES  9                     // 144 / 16
#define SPLITK  8
#define KSEG    (N_PTS / SPLITK)      // 2048 n-values per K segment
#define NCHUNK  (N_PTS / 32)          // 512 K-chunks of 32
#define EPSV    1e-8f

// workspace layout (float-sized slots)
#define SWY_OFF   0
#define S1_OFF    (SWY_OFF + FK * C_DIM)    // 262144
#define MASS_OFF  (S1_OFF + FK)             // 264192
#define LOSS_OFF  (MASS_OFF + FK)           // 266240
#define YSQ_OFF   (LOSS_OFF + 8)            // 266248
#define BPK_OFF   (YSQ_OFF + N_PTS)         // 282632 (32B aligned: *4 % 32 == 0)
#define WS_ZERO_FLOATS YSQ_OFF
// bpack: NCHUNK * NTILES * 32 lanes * 8 uints = 1,179,648 uints (4.5 MiB)

__device__ __forceinline__ unsigned pk_bf16(float lo, float hi) {
    bf16x2 v;
    v[0] = (__bf16)lo;
    v[1] = (__bf16)hi;
    return __builtin_bit_cast(unsigned, v);
}

// ---------------- y_sq[n] = sum_c teacher[n,c]^2 : one wave per n ----------
__global__ __launch_bounds__(256)
void ysq_kernel(const float* __restrict__ teacher, float* __restrict__ ysq) {
    int wave = (int)((blockIdx.x * blockDim.x + threadIdx.x) >> 5);
    int lane = (int)(threadIdx.x & 31);
    if (wave >= N_PTS) return;
    const float4* row = (const float4*)(teacher + (size_t)wave * C_DIM);
    float4 v = row[lane];
    float s = v.x * v.x + v.y * v.y + v.z * v.z + v.w * v.w;
    for (int off = 16; off; off >>= 1) s += __shfl_xor(s, off, 32);
    if (lane == 0) ysq[wave] = s;
}

// ------- pre-pack B = [teacher | ysq | 1 | 0] into WMMA fragment image -----
// bpack[((chunk*NTILES + tile)*32 + lane)*8 + i] = packed bf16 K-pair
// lane 0-15: col = tile*16 + lane, K = 0..15 ; lane 16-31: same col, K = 16..31
__global__ __launch_bounds__(128)
void bpack_kernel(const float* __restrict__ teacher,
                  const float* __restrict__ ysq,
                  unsigned* __restrict__ bpack) {
    int gid = (int)(blockIdx.x * blockDim.x + threadIdx.x);
    if (gid >= NCHUNK * NTILES * 32) return;
    int L   = gid & 31;
    int j   = (gid >> 5) % NTILES;
    int t   = gid / (32 * NTILES);
    int col = j * 16 + (L & 15);
    int n0  = t * 32 + ((L < 16) ? 0 : 16);

    unsigned o[8];
    for (int i = 0; i < 8; ++i) {
        int nlo = n0 + 2 * i, nhi = nlo + 1;
        float lo, hi;
        if (col < C_DIM)      { lo = teacher[(size_t)nlo * C_DIM + col];
                                hi = teacher[(size_t)nhi * C_DIM + col]; }
        else if (col == 128)  { lo = ysq[nlo]; hi = ysq[nhi]; }
        else if (col == 129)  { lo = 1.0f;     hi = 1.0f; }
        else                  { lo = 0.0f;     hi = 0.0f; }
        o[i] = pk_bf16(lo, hi);
    }
    unsigned* dst = bpack + (size_t)gid * 8;
    ((uint4*)dst)[0] = make_uint4(o[0], o[1], o[2], o[3]);
    ((uint4*)dst)[1] = make_uint4(o[4], o[5], o[6], o[7]);
}

// ---------------- main fused GEMM: membership^T @ [teacher | ysq | 1] ------
__global__ __launch_bounds__(128)
void gemm_kernel(const float* __restrict__ mem,
                 const unsigned* __restrict__ bpack,
                 float* __restrict__ ws) {
    // A tile per wave, packed bf16 K-pairs: [m 0..15][n-pair 0..15], stride 17
    __shared__ unsigned lds_apk[4][16][17];

    float* swy  = ws + SWY_OFF;
    float* s1   = ws + S1_OFF;
    float* mass = ws + MASS_OFF;

    const int tid    = (int)threadIdx.x;
    const int w      = tid >> 5;
    const int lane   = tid & 31;
    const int fkBase = (int)blockIdx.x * 64;          // 4 waves x 16 rows
    const int nSeg0  = (int)blockIdx.y * KSEG;

    v8f acc[NTILES];
    for (int j = 0; j < NTILES; ++j)
        for (int i = 0; i < 8; ++i) acc[j][i] = 0.0f;

    // A staging role: row-pair pA (n = 2*pA, 2*pA+1), fk-half hA (8 cols)
    const int pA = lane & 15;
    const int hA = lane >> 4;
    // A fragment role: row m, K-half -> pair offset 0 or 8
    const int mL = lane & 15;
    const int kp = (lane < 16) ? 0 : 8;

    for (int n0 = nSeg0; n0 < nSeg0 + KSEG; n0 += 32) {
        // ---- stage A: 2 rows x 8 fk per lane, nontemporal b128 streams
        {
            const float* a0 =
                mem + (size_t)(n0 + 2 * pA) * FK + fkBase + w * 16 + hA * 8;
            const float* a1 = a0 + FK;
            v4f r0a = __builtin_nontemporal_load((const v4f*)a0);
            v4f r0b = __builtin_nontemporal_load(((const v4f*)a0) + 1);
            v4f r1a = __builtin_nontemporal_load((const v4f*)a1);
            v4f r1b = __builtin_nontemporal_load(((const v4f*)a1) + 1);
            if (n0 + 32 < nSeg0 + KSEG)
                __builtin_prefetch(a0 + (size_t)32 * FK, 0, 1);
            int mB = hA * 8;
            lds_apk[w][mB + 0][pA] = pk_bf16(r0a.x, r1a.x);
            lds_apk[w][mB + 1][pA] = pk_bf16(r0a.y, r1a.y);
            lds_apk[w][mB + 2][pA] = pk_bf16(r0a.z, r1a.z);
            lds_apk[w][mB + 3][pA] = pk_bf16(r0a.w, r1a.w);
            lds_apk[w][mB + 4][pA] = pk_bf16(r0b.x, r1b.x);
            lds_apk[w][mB + 5][pA] = pk_bf16(r0b.y, r1b.y);
            lds_apk[w][mB + 6][pA] = pk_bf16(r0b.z, r1b.z);
            lds_apk[w][mB + 7][pA] = pk_bf16(r0b.w, r1b.w);
        }
        __syncthreads();   // intra-wave DS ordering insurance (4 waves, cheap)

        // ---- A fragment: one contiguous 8-uint run
        v8u au;
        {
            const unsigned* arow = &lds_apk[w][mL][0];
            au[0] = arow[kp + 0];
            au[1] = arow[kp + 1];
            au[2] = arow[kp + 2];
            au[3] = arow[kp + 3];
            au[4] = arow[kp + 4];
            au[5] = arow[kp + 5];
            au[6] = arow[kp + 6];
            au[7] = arow[kp + 7];
        }
        v16bf afv = __builtin_bit_cast(v16bf, au);

        // ---- B fragments: per-lane contiguous 32B, coalesced across lanes
        const unsigned* bbase =
            bpack + ((size_t)(n0 >> 5) * NTILES * 32 + lane) * 8;
        for (int j = 0; j < NTILES; ++j) {
            v8u bu = *(const v8u*)(bbase + (size_t)j * 32 * 8);
            v16bf bfv = __builtin_bit_cast(v16bf, bu);
            acc[j] = __builtin_amdgcn_wmma_f32_16x16x32_bf16(
                false, afv, false, bfv, (short)0, acc[j], false, false);
        }
        __syncthreads();
    }

    // ---- split-K accumulate (C/D layout: VGPR r -> M = r (+8 for hi lanes))
    for (int j = 0; j < NTILES; ++j) {
        for (int r = 0; r < 8; ++r) {
            float v  = acc[j][r];
            int   M  = r + ((lane < 16) ? 0 : 8);
            int   fk = fkBase + w * 16 + M;
            int   col = j * 16 + (lane & 15);
            if (col < C_DIM)       atomicAdd(&swy[(size_t)fk * C_DIM + col], v);
            else if (col == 128)   atomicAdd(&s1[fk], v);
            else if (col == 129)   atomicAdd(&mass[fk], v);
        }
    }
}

// ---------------- per-feature epilogue ------------------------------------
__global__ __launch_bounds__(128)
void epilogue_kernel(float* __restrict__ ws) {
    __shared__ float cent[K_BINS][C_DIM];     // 16 KB
    __shared__ float csq[K_BINS], bmsh[K_BINS], s1sh[K_BINS];
    __shared__ float red[4][128];

    const float* swy  = ws + SWY_OFF;
    const float* s1   = ws + S1_OFF;
    const float* mass = ws + MASS_OFF;
    float*       loss = ws + LOSS_OFF;

    const int f = (int)blockIdx.x;
    const int tid = (int)threadIdx.x;

    if (tid < K_BINS) {
        bmsh[tid] = mass[f * K_BINS + tid] + EPSV;   // bin_mass = sum + eps
        s1sh[tid] = s1[f * K_BINS + tid];
    }
    __syncthreads();
    for (int idx = tid; idx < K_BINS * C_DIM; idx += 128) {
        int k = idx >> 7, c = idx & 127;
        cent[k][c] = swy[(size_t)(f * K_BINS + k) * C_DIM + c] / bmsh[k];
    }
    __syncthreads();
    if (tid < K_BINS) {
        float s = 0.f;
        for (int c = 0; c < C_DIM; ++c) { float x = cent[tid][c]; s += x * x; }
        csq[tid] = s;
    }
    __syncthreads();

    float disp = 0.f, ent = 0.f, rep = 0.f, inter = 0.f;
    if (tid < K_BINS) {
        float bm = bmsh[tid], summ = bm - EPSV, cs = csq[tid];
        // sum_n m*(y_sq + c_sq - 2*cross)/bin_mass ; sum_n m*cross = bm*c_sq
        disp = (s1sh[tid] + summ * cs - 2.0f * bm * cs) / bm;
        float p = bm / (float)N_PTS;
        ent  = p * logf(p + EPSV);
    }
    for (int flat = tid; flat < K_BINS * K_BINS; flat += 128) {
        int k = flat >> 5, j = flat & 31;
        if (j > k) {
            float dot = 0.f;
            for (int c = 0; c < C_DIM; ++c) dot += cent[k][c] * cent[j][c];
            float d = csq[k] + csq[j] - 2.0f * dot;     // TEMPERATURE = 1
            inter += __expf(-d);
            if (j == k + 1) rep += __expf(-d);
        }
    }
    red[0][tid] = disp; red[1][tid] = ent; red[2][tid] = rep; red[3][tid] = inter;
    __syncthreads();
    if (tid == 0) {
        float a = 0.f, b = 0.f, c = 0.f, d = 0.f;
        for (int i = 0; i < 128; ++i) {
            a += red[0][i]; b += red[1][i]; c += red[2][i]; d += red[3][i];
        }
        atomicAdd(&loss[0], a); atomicAdd(&loss[1], b);
        atomicAdd(&loss[2], c); atomicAdd(&loss[3], d);
    }
}

// ---------------- final combine (writes all 5 outputs fresh) ---------------
__global__ void combine_kernel(const float* __restrict__ ws, float* __restrict__ out) {
    if (threadIdx.x == 0 && blockIdx.x == 0) {
        const float* loss = ws + LOSS_OFF;
        float disp  = loss[0];
        float ent   = loss[1];
        float rep   = loss[2];
        float inter = loss[3] / (float)F_DIM;
        out[0] = disp + 0.1f * ent + 0.5f * rep + 0.3f * inter;
        out[1] = disp;
        out[2] = ent;
        out[3] = rep;
        out[4] = inter;
    }
}

extern "C" void kernel_launch(void* const* d_in, const int* in_sizes, int n_in,
                              void* d_out, int out_size, void* d_ws, size_t ws_size,
                              hipStream_t stream) {
    const float* mem     = (const float*)d_in[0];   // (16384, 64, 32) f32
    const float* teacher = (const float*)d_in[1];   // (16384, 128)    f32
    float*    ws    = (float*)d_ws;
    unsigned* bpack = (unsigned*)(ws + BPK_OFF);
    float*    out   = (float*)d_out;

    // zero split-K + loss accumulators (graph-capturable)
    (void)hipMemsetAsync(d_ws, 0, (size_t)WS_ZERO_FLOATS * sizeof(float), stream);

    ysq_kernel<<<dim3(N_PTS / 8), dim3(256), 0, stream>>>(teacher, ws + YSQ_OFF);
    bpack_kernel<<<dim3((NCHUNK * NTILES * 32 + 127) / 128), dim3(128), 0, stream>>>(
        teacher, ws + YSQ_OFF, bpack);
    gemm_kernel<<<dim3(FK / 64, SPLITK), dim3(128), 0, stream>>>(mem, bpack, ws);
    epilogue_kernel<<<dim3(F_DIM), dim3(128), 0, stream>>>(ws);
    combine_kernel<<<dim3(1), dim3(32), 0, stream>>>(ws, out);
}
